// nanoGPT_5927054868596
// MI455X (gfx1250) — matrix operand order repaired and checked by reading, not compiled, not run on hardware
//
#include <hip/hip_runtime.h>
#include <math.h>

#define DMODEL 768
#define NH 12
#define HD 64
#define NLAYER 4
#define SEQ 1024
#define BATCH 2
#define MROWS (BATCH*SEQ)   /* 2048 */
#define DFF 3072
#define VOCAB 32000
#define LMCHUNK 4000
#define EPS 1e-5f

typedef __attribute__((ext_vector_type(16))) _Float16 v16h;
typedef __attribute__((ext_vector_type(8)))  float    v8f;

union HalfVec { uint4 u[2]; v16h v; _Float16 h[16]; };

#if __has_builtin(__builtin_amdgcn_sched_barrier)
#define SCHED_FENCE() __builtin_amdgcn_sched_barrier(0)
#else
#define SCHED_FENCE() asm volatile("" ::: "memory")
#endif

// ---- gfx1250 async global->LDS copy (ASYNCcnt) --------------------------
__device__ __forceinline__ void async_b128(unsigned lds_off, const _Float16* g) {
  asm volatile("global_load_async_to_lds_b128 %0, %1, off" :: "v"(lds_off), "v"(g) : "memory");
}
__device__ __forceinline__ void async_b64(unsigned lds_off, const _Float16* g) {
  asm volatile("global_load_async_to_lds_b64 %0, %1, off" :: "v"(lds_off), "v"(g) : "memory");
}
__device__ __forceinline__ void wait_async0() {   // drain all async copies
  asm volatile("s_wait_asynccnt 0x0" ::: "memory");
}
__device__ __forceinline__ void wait_async1() {   // oldest copy done, 1 may fly
  asm volatile("s_wait_asynccnt 0x1" ::: "memory");
}
__device__ __forceinline__ unsigned lds_addr(const void* p) {
  // addrspace(3) offset == low 32 bits of the generic pointer
  return (unsigned)(unsigned long long)p;
}

// ---------------------------------------------------------------- embedding
__global__ void embed_kernel(const int* __restrict__ idx, const float* __restrict__ wte,
                             const float* __restrict__ wpe, float* __restrict__ x) {
  int row = blockIdx.x;                 // 0..MROWS-1  (b*SEQ + t)
  int t   = row % SEQ;
  int tok = idx[row];
  const float* we = wte + (size_t)tok * DMODEL;
  const float* pe = wpe + (size_t)t   * DMODEL;
  float* xr = x + (size_t)row * DMODEL;
  for (int c = threadIdx.x; c < DMODEL; c += blockDim.x)
    xr[c] = (we[c] + pe[c]) * 27.712812921102035f;   // sqrt(768)
}

// ------------------------------------------------- DynamicTanh -> f16 stage
__global__ void dyt_kernel(const float* __restrict__ x, const float* __restrict__ alpha,
                           const float* __restrict__ w, const float* __restrict__ b,
                           _Float16* __restrict__ out) {
  __shared__ float sh1[256], sh2[256];
  int row = blockIdx.x;
  const float* xr = x + (size_t)row * DMODEL;
  float s1 = 0.f, s2 = 0.f;
  for (int c = threadIdx.x; c < DMODEL; c += blockDim.x) { float t = xr[c]; s1 += t; s2 += t*t; }
  sh1[threadIdx.x] = s1; sh2[threadIdx.x] = s2; __syncthreads();
  for (int s = 128; s > 0; s >>= 1) {
    if ((int)threadIdx.x < s) { sh1[threadIdx.x] += sh1[threadIdx.x+s]; sh2[threadIdx.x] += sh2[threadIdx.x+s]; }
    __syncthreads();
  }
  float mean = sh1[0] * (1.f/DMODEL);
  float var  = sh2[0] * (1.f/DMODEL) - mean*mean;
  float rstd = 1.f / sqrtf(var + EPS);
  float a = alpha[0];
  _Float16* orow = out + (size_t)row * DMODEL;
  for (int c = threadIdx.x; c < DMODEL; c += blockDim.x) {
    float xn = (xr[c] - mean) * rstd;
    orow[c] = (_Float16)(w[c] * tanhf(a * xn) + b[c]);
  }
}

// ------------------- pack fp32 weight [K,N] into strip-major WMMA B frags
// Tile (nt,kt): strip s = nt/NTpack, j = nt%NTpack, stored at
// ((s*Ktiles + kt)*NTpack + j)*512.  One (strip,kt) group = NTpack*1KB
// contiguous -> whole-block async copy.  Within a tile, lane l holds 16
// contiguous halves: B[kt*32 + ((l&16)?16:0) + i][nt*16 + (l&15)].
__global__ void pack_w_kernel(const float* __restrict__ W, _Float16* __restrict__ dst,
                              int Ktiles, int NTpack, long sk, long sn) {
  int gid  = blockIdx.x * blockDim.x + threadIdx.x;
  int lane = gid & 31;
  int tile = gid >> 5;
  int nt = tile / Ktiles;
  int kt = tile - nt * Ktiles;
  int s  = nt / NTpack;
  int j  = nt - s * NTpack;
  int n  = nt * 16 + (lane & 15);
  int kb = kt * 32 + ((lane & 16) ? 16 : 0);
  HalfVec hv;
#pragma unroll
  for (int i = 0; i < 16; ++i) hv.h[i] = (_Float16)W[(long)(kb + i) * sk + (long)n * sn];
  size_t toff = ((size_t)(s * Ktiles + kt) * NTpack + j) * 512;
  uint4* d = (uint4*)(dst + toff + lane * 16);
  d[0] = hv.u[0]; d[1] = hv.u[1];
}

// ----------------------------------------------- WMMA GEMM C = A(f16) @ Bp
// Block = 8 waves: all share one 16*NT-column strip.  B tiles staged in LDS
// with a 3-deep async pipeline (2 copies in flight, wait asynccnt<=1);
// A fragments register double-buffered; B fragments batch-loaded from LDS
// into distinct registers (sched fences) so DScnt waits interleave with WMMAs.
template<int NT>
__global__ __launch_bounds__(256)
void gemm_wmma_kernel(const _Float16* __restrict__ A, const _Float16* __restrict__ Bp,
                      float* __restrict__ C, int K, int ldc) {
  __shared__ _Float16 smem[3][NT * 512];
  const int tid   = threadIdx.x;
  const int lane  = tid & 31;
  const int wv    = tid >> 5;                 // 0..7
  const int strip = blockIdx.x;
  const int mt    = blockIdx.y * 8 + wv;
  const int Ktiles = K >> 5;

  const int mrow  = mt * 16 + (lane & 15);
  const int kbase = (lane & 16) ? 16 : 0;     // A: hi half-wave holds K+16..31
  const _Float16* pa = A + (size_t)mrow * K + kbase;

  const int chunkHalves = NT * 512;           // halves per (strip,kt) group
  const int perThread   = chunkHalves / 256;  // 8 (NT=4) or 4 (NT=2)
  const _Float16* pbStrip = Bp + (size_t)strip * Ktiles * chunkHalves;

  v8f acc[NT];
#pragma unroll
  for (int j = 0; j < NT; ++j) acc[j] = (v8f){0.f,0.f,0.f,0.f,0.f,0.f,0.f,0.f};

  // prologue: stage copies for kt=0 and kt=1; load A(0)
  {
    const _Float16* s0 = pbStrip + tid * perThread;
    unsigned d0 = lds_addr(&smem[0][tid * perThread]);
    if (NT == 4) async_b128(d0, s0); else async_b64(d0, s0);
    if (Ktiles > 1) {
      const _Float16* s1 = pbStrip + (size_t)chunkHalves + tid * perThread;
      unsigned d1 = lds_addr(&smem[1][tid * perThread]);
      if (NT == 4) async_b128(d1, s1); else async_b64(d1, s1);
    }
  }
  HalfVec avCur;
  avCur.u[0] = *(const uint4*)(pa);
  avCur.u[1] = *(const uint4*)(pa + 8);
  if (Ktiles > 1) wait_async1(); else wait_async0();
  __syncthreads();

  int buf = 0;                                // kt % 3, maintained by rotation
  for (int kt = 0; kt < Ktiles; ++kt) {
    bool more  = (kt + 1 < Ktiles);
    bool more2 = (kt + 2 < Ktiles);
    int nbuf = (buf == 2) ? 0 : buf + 1;
    int fbuf = (nbuf == 2) ? 0 : nbuf + 1;    // (kt+2) % 3
    if (more2) {
      // buffer fbuf was last read at iteration kt-1 (barrier passed) -> safe
      const _Float16* src = pbStrip + (size_t)(kt + 2) * chunkHalves + tid * perThread;
      unsigned dst = lds_addr(&smem[fbuf][tid * perThread]);
      if (NT == 4) async_b128(dst, src); else async_b64(dst, src);
      if (kt + 3 < Ktiles)
        __builtin_prefetch(pbStrip + (size_t)(kt + 3) * chunkHalves + tid * perThread, 0, 3);
    }
    HalfVec avNext;
    if (more) {                               // register double-buffer A
      avNext.u[0] = *(const uint4*)(pa + (size_t)(kt + 1) * 32);
      avNext.u[1] = *(const uint4*)(pa + (size_t)(kt + 1) * 32 + 8);
    }
    // batch all B-fragment LDS reads into distinct registers, THEN the WMMAs.
    // Fences keep the scheduler from sinking each ds_load next to its WMMA
    // (which coalesces the fragments into one register set and forces
    // s_wait_dscnt 0x0 before every WMMA).
    const _Float16* sb = &smem[buf][0];
    HalfVec bv[NT];
#pragma unroll
    for (int j = 0; j < NT; ++j) {
      const uint4* pbu = (const uint4*)(sb + j * 512 + lane * 16);
      bv[j].u[0] = pbu[0]; bv[j].u[1] = pbu[1];
    }
    SCHED_FENCE();
#pragma unroll
    for (int j = 0; j < NT; ++j)
      acc[j] = __builtin_amdgcn_wmma_f32_16x16x32_f16(
          false, avCur.v, false, bv[j].v, (short)0, acc[j], false, false);
    SCHED_FENCE();
    if (more) {
      avCur = avNext;
      if (more2) wait_async1(); else wait_async0();  // oldest copy (kt+1) landed
      __syncthreads();                               // block-wide visibility
    }
    buf = nbuf;
  }

  // C/D layout: VGPR i holds M = i + (lane>=16 ? 8 : 0), N = lane&15
  int r0   = mt * 16 + ((lane & 16) ? 8 : 0);
  int col0 = lane & 15;
#pragma unroll
  for (int j = 0; j < NT; ++j) {
    int col = (strip * NT + j) * 16 + col0;
#pragma unroll
    for (int i = 0; i < 8; ++i) C[(size_t)(r0 + i) * ldc + col] = acc[j][i];
  }
}

// ----------------------------------------- causal attention, online softmax
__global__ void attn_kernel(const float* __restrict__ q, const float* __restrict__ k,
                            const float* __restrict__ v, float* __restrict__ o) {
  int gid = blockIdx.x * blockDim.x + threadIdx.x;   // (b*NH + h)*SEQ + t
  if (gid >= BATCH * NH * SEQ) return;
  int t  = gid % SEQ;
  int bh = gid / SEQ;
  int h  = bh % NH;
  int b  = bh / NH;
  const float* qr = q + ((size_t)(b * SEQ + t) * DMODEL) + h * HD;
  float qv[HD];
#pragma unroll
  for (int d = 0; d < HD; ++d) qv[d] = qr[d] * 0.125f;   // 1/sqrt(64)
  float m = -3.4e38f, l = 0.f;
  float accv[HD];
#pragma unroll
  for (int d = 0; d < HD; ++d) accv[d] = 0.f;
  const float* kb = k + (size_t)b * SEQ * DMODEL + h * HD;
  const float* vb = v + (size_t)b * SEQ * DMODEL + h * HD;
  for (int j = 0; j <= t; ++j) {
    const float* kr = kb + (size_t)j * DMODEL;
    float s = 0.f;
#pragma unroll
    for (int d = 0; d < HD; ++d) s += qv[d] * kr[d];
    float mn   = fmaxf(m, s);
    float corr = expf(m - mn);
    float p    = expf(s - mn);
    l = l * corr + p;
    const float* vr = vb + (size_t)j * DMODEL;
#pragma unroll
    for (int d = 0; d < HD; ++d) accv[d] = accv[d] * corr + p * vr[d];
    m = mn;
  }
  float inv = 1.f / l;
  float* orow = o + ((size_t)(b * SEQ + t) * DMODEL) + h * HD;
#pragma unroll
  for (int d = 0; d < HD; ++d) orow[d] = accv[d] * inv;
}

// -------------------------------------------------------------- elementwise
__global__ void resid_kernel(float* __restrict__ x, const float* __restrict__ t,
                             const float* __restrict__ bias, int ncols) {
  int i = blockIdx.x * blockDim.x + threadIdx.x;
  int c = i % ncols;
  x[i] += t[i] + bias[c];
}

__global__ void gelu_kernel(const float* __restrict__ fc, const float* __restrict__ bias,
                            _Float16* __restrict__ out) {
  int i = blockIdx.x * blockDim.x + threadIdx.x;
  int c = i % DFF;
  float t = fc[i] + bias[c];
  float g = 0.5f * t * (1.f + erff(t * 0.70710678118654752f));   // exact GELU
  out[i] = (_Float16)g;
}

__global__ void f32_to_f16_kernel(const float* __restrict__ in, _Float16* __restrict__ out, int n) {
  int i = blockIdx.x * blockDim.x + threadIdx.x;
  if (i < n) out[i] = (_Float16)in[i];
}

// ---------------------------------------------------------------- launcher
extern "C" void kernel_launch(void* const* d_in, const int* in_sizes, int n_in,
                              void* d_out, int out_size, void* d_ws, size_t ws_size,
                              hipStream_t stream) {
  const int*   idx       = (const int*)  d_in[0];
  const float* wte       = (const float*)d_in[1];
  const float* wpe       = (const float*)d_in[2];
  const float* ln1_alpha = (const float*)d_in[3];
  const float* ln1_w     = (const float*)d_in[4];
  const float* ln1_b     = (const float*)d_in[5];
  const float* wq        = (const float*)d_in[6];
  const float* wk        = (const float*)d_in[7];
  const float* wv        = (const float*)d_in[8];
  const float* wo        = (const float*)d_in[9];
  const float* bo        = (const float*)d_in[10];
  const float* ln2_alpha = (const float*)d_in[11];
  const float* ln2_w     = (const float*)d_in[12];
  const float* ln2_b     = (const float*)d_in[13];
  const float* w_fc      = (const float*)d_in[14];
  const float* b_fc      = (const float*)d_in[15];
  const float* w_proj    = (const float*)d_in[16];
  const float* b_proj    = (const float*)d_in[17];
  const float* lnf_alpha = (const float*)d_in[18];
  const float* lnf_w     = (const float*)d_in[19];
  const float* lnf_b     = (const float*)d_in[20];
  float* out = (float*)d_out;

  char* ws = (char*)d_ws;
  size_t off = 0;
  auto alloc = [&](size_t bytes) -> void* {
    void* p = ws + off;
    off += (bytes + 255) & ~(size_t)255;
    return p;
  };
  float*    x    = (float*)   alloc((size_t)MROWS * DMODEL * 4);
  float*    qb   = (float*)   alloc((size_t)MROWS * DMODEL * 4);
  float*    kbuf = (float*)   alloc((size_t)MROWS * DMODEL * 4);
  float*    vbuf = (float*)   alloc((size_t)MROWS * DMODEL * 4);
  float*    ob   = (float*)   alloc((size_t)MROWS * DMODEL * 4);
  float*    tmp2 = (float*)   alloc((size_t)MROWS * DMODEL * 4);
  float*    tmp  = (float*)   alloc((size_t)MROWS * DFF    * 4);
  _Float16* hb1  = (_Float16*)alloc((size_t)MROWS * DMODEL * 2);
  _Float16* hb2  = (_Float16*)alloc((size_t)MROWS * DFF    * 2);
  _Float16* wpack= (_Float16*)alloc((size_t)DMODEL * LMCHUNK * 2);  // max packed chunk

  auto pack = [&](const float* W, long sk, long sn, int K, int N) {
    int NTpack = (N % 64 == 0) ? 4 : 2;
    int tiles = (K / 32) * (N / 16);
    pack_w_kernel<<<tiles * 32 / 256, 256, 0, stream>>>(W, wpack, K / 32, NTpack, sk, sn);
  };
  auto gemm = [&](const _Float16* A, float* C, int M, int N, int K, int ldc) {
    if (N % 64 == 0) {
      dim3 grid(N / 64, M / 128);
      gemm_wmma_kernel<4><<<grid, 256, 0, stream>>>(A, wpack, C, K, ldc);
    } else {
      dim3 grid(N / 32, M / 128);
      gemm_wmma_kernel<2><<<grid, 256, 0, stream>>>(A, wpack, C, K, ldc);
    }
  };

  embed_kernel<<<MROWS, 256, 0, stream>>>(idx, wte, wpe, x);

  for (int l = 0; l < NLAYER; ++l) {
    // ---- attention block
    dyt_kernel<<<MROWS, 256, 0, stream>>>(x, ln1_alpha + l, ln1_w + (size_t)l*DMODEL,
                                          ln1_b + (size_t)l*DMODEL, hb1);
    pack(wq + (size_t)l*DMODEL*DMODEL, DMODEL, 1, DMODEL, DMODEL);
    gemm(hb1, qb,   MROWS, DMODEL, DMODEL, DMODEL);
    pack(wk + (size_t)l*DMODEL*DMODEL, DMODEL, 1, DMODEL, DMODEL);
    gemm(hb1, kbuf, MROWS, DMODEL, DMODEL, DMODEL);
    pack(wv + (size_t)l*DMODEL*DMODEL, DMODEL, 1, DMODEL, DMODEL);
    gemm(hb1, vbuf, MROWS, DMODEL, DMODEL, DMODEL);
    attn_kernel<<<(BATCH * NH * SEQ) / 128, 128, 0, stream>>>(qb, kbuf, vbuf, ob);
    f32_to_f16_kernel<<<(MROWS * DMODEL) / 256, 256, 0, stream>>>(ob, hb1, MROWS * DMODEL);
    pack(wo + (size_t)l*DMODEL*DMODEL, DMODEL, 1, DMODEL, DMODEL);
    gemm(hb1, tmp2, MROWS, DMODEL, DMODEL, DMODEL);
    resid_kernel<<<(MROWS * DMODEL) / 256, 256, 0, stream>>>(x, tmp2, bo + (size_t)l*DMODEL, DMODEL);
    // ---- MLP block
    dyt_kernel<<<MROWS, 256, 0, stream>>>(x, ln2_alpha + l, ln2_w + (size_t)l*DMODEL,
                                          ln2_b + (size_t)l*DMODEL, hb1);
    pack(w_fc + (size_t)l*DMODEL*DFF, DFF, 1, DMODEL, DFF);
    gemm(hb1, tmp, MROWS, DFF, DMODEL, DFF);
    gelu_kernel<<<(MROWS * DFF) / 256, 256, 0, stream>>>(tmp, b_fc + (size_t)l*DFF, hb2);
    pack(w_proj + (size_t)l*DFF*DMODEL, DMODEL, 1, DFF, DMODEL);
    gemm(hb2, tmp2, MROWS, DMODEL, DFF, DMODEL);
    resid_kernel<<<(MROWS * DMODEL) / 256, 256, 0, stream>>>(x, tmp2, b_proj + (size_t)l*DMODEL, DMODEL);
  }

  // ---- final DyT + tied lm_head (wte^T), in column chunks to bound scratch
  dyt_kernel<<<MROWS, 256, 0, stream>>>(x, lnf_alpha, lnf_w, lnf_b, hb1);
  for (int c = 0; c < VOCAB / LMCHUNK; ++c) {
    // B[k][n] = wte[(c*LMCHUNK + n)*DMODEL + k]  ->  sk = 1, sn = DMODEL
    pack(wte + (size_t)c * LMCHUNK * DMODEL, 1, DMODEL, DMODEL, LMCHUNK);
    gemm(hb1, out + (size_t)c * LMCHUNK, MROWS, LMCHUNK, DMODEL, VOCAB);
  }
}
